// Attention_31791347925834
// MI455X (gfx1250) — matrix operand order repaired and checked by reading, not compiled, not run on hardware
//
#include <hip/hip_runtime.h>
#include <hip/hip_bf16.h>

#define DIM        2048
#define N_HEADS    32
#define N_KV_HEADS 8
#define HEAD_DIM   64
#define N_REP      (N_HEADS / N_KV_HEADS)
#define BSZ        2
#define SEQLEN     2048
#define KV_DIM     (N_KV_HEADS * HEAD_DIM)   // 512

typedef __bf16 bf16_t;
typedef bf16_t bf16x4  __attribute__((ext_vector_type(4)));
typedef bf16_t bf16x8  __attribute__((ext_vector_type(8)));
typedef bf16_t bf16x16 __attribute__((ext_vector_type(16)));
typedef float  floatx4 __attribute__((ext_vector_type(4)));
typedef float  floatx8 __attribute__((ext_vector_type(8)));
typedef unsigned int uint32x4 __attribute__((ext_vector_type(4)));
typedef int    int32x4 __attribute__((ext_vector_type(4)));
typedef int    int32x8 __attribute__((ext_vector_type(8)));

static __device__ __forceinline__ floatx8 wmma_bf16(bf16x16 a, bf16x16 b, floatx8 c) {
    return __builtin_amdgcn_wmma_f32_16x16x32_bf16(
        /*neg_a=*/false, a, /*neg_b=*/false, b,
        /*c_mod=*/(short)0, c, /*reuse_a=*/false, /*reuse_b=*/false);
}

// Async copy 16 bytes global -> LDS (no VGPR data path, tracked by ASYNCcnt).
static __device__ __forceinline__ void async_load_b128(unsigned lds_off,
                                                       unsigned long long gaddr) {
    asm volatile("global_load_async_to_lds_b128 %0, %1, off"
                 :: "v"(lds_off), "v"(gaddr) : "memory");
}
static __device__ __forceinline__ void wait_asynccnt0() {
    asm volatile("s_wait_asynccnt 0" ::: "memory");
}

// ---------------------------------------------------------------------------
// Tensor Data Mover: 2D tile load (rows x 64 bf16 elements) into LDS.
// D# per CDNA5 ISA: group0 = {count, lds_addr,
// global_addr, type=2}; group1 = {data_size=2B, tensor/tile dims, stride}.
// ---------------------------------------------------------------------------
#if defined(__has_builtin)
#  if __has_builtin(__builtin_amdgcn_tensor_load_to_lds)
#    define HAVE_TDM 1
#  endif
#endif

#ifdef HAVE_TDM
static __device__ __forceinline__ void tdm_load_tile_64x64_bf16(
        unsigned lds_addr, unsigned long long gaddr, unsigned row_stride_elems) {
    uint32x4 g0;
    g0[0] = 1u;                                         // count=1 valid descriptor
    g0[1] = lds_addr;                                   // lds_addr [63:32]
    g0[2] = (unsigned)(gaddr & 0xffffffffu);            // global_addr low
    g0[3] = (unsigned)((gaddr >> 32) & 0x01ffffffu)     // global_addr [56:32]
          | (2u << 30);                                 // type=2 ("image")
    int32x8 g1;
    g1[0] = 0x00010000;                                 // data_size = 2 bytes
    g1[1] = (int)(64u << 16);                           // tensor_dim0 = 64 (lo16)
    g1[2] = 0;                                          // dim0 hi16=0, dim1 lo16=0
    g1[3] = (int)(1u | (64u << 16));                    // dim1=65536, tile_dim0=64
    g1[4] = 64;                                         // tile_dim1=64, tile_dim2=0
    g1[5] = (int)row_stride_elems;                      // tensor_dim0_stride lo32
    g1[6] = 0;
    g1[7] = 0;
    int32x4 z4 = (int32x4)0;
#if __has_include(<hip/amd_detail/amd_gfx1250_TDM.h>)
    int32x8 z8 = (int32x8)0;                            // 6-arg therock variant
    __builtin_amdgcn_tensor_load_to_lds(g0, g1, z4, z4, z8, 0);
#else
    __builtin_amdgcn_tensor_load_to_lds(g0, g1, z4, z4, 0);  // 5-arg ROCm variant
#endif
}
#endif

// ---------------------------------------------------------------------------
// fp32 -> bf16 one-shot conversion (amortized: weights reused 32x, x reused 24x)
// ---------------------------------------------------------------------------
__global__ void cvt_f32_to_bf16(const float* __restrict__ in,
                                bf16_t* __restrict__ out, int n) {
    const int i = (blockIdx.x * blockDim.x + threadIdx.x) * 4;
    if (i >= n) return;
    floatx4 v = *(const floatx4*)(in + i);
    bf16x4 o;
#pragma unroll
    for (int j = 0; j < 4; ++j) o[j] = (bf16_t)v[j];
    *(bf16x4*)(out + i) = o;
}

// ---------------------------------------------------------------------------
// GEMM: C[M,N] = A[M,K] * B[N,K]^T, A/B bf16, C fp32 or bf16.
// 128x128x32 block tile, 8 waves, wave tile 32x64.  Tiles staged with
// double-buffered global_load_async_to_lds_b128 (DMA overlaps WMMA).
// ---------------------------------------------------------------------------
#define GB_M 128
#define GB_N 128
#define GB_K 32

__global__ __launch_bounds__(256)
void gemm_bf16(const bf16_t* __restrict__ A, const bf16_t* __restrict__ B,
               void* __restrict__ Cout, int M, int N, int K, int store_bf16) {
    __shared__ __align__(32) bf16_t lA[2][GB_M * GB_K];
    __shared__ __align__(32) bf16_t lB[2][GB_M * GB_K];

    const int tid  = threadIdx.x;
    const int lane = tid & 31;
    const int wid  = tid >> 5;
    const int hi   = lane >> 4;
    const int lo   = lane & 15;

    const int bm = blockIdx.y * GB_M;
    const int bn = blockIdx.x * GB_N;
    const int wm = (wid & 3) * 32;
    const int wn = (wid >> 2) * 64;

    floatx8 acc[2][4];
#pragma unroll
    for (int i = 0; i < 2; ++i)
#pragma unroll
        for (int j = 0; j < 4; ++j) acc[i][j] = (floatx8)0.0f;

    // Stage a 128x32 bf16 panel of A and B into LDS buffer `buf` via async DMA.
    auto stage = [&](int k0, int buf) {
#pragma unroll
        for (int it = 0; it < 2; ++it) {
            const int idx = tid + it * 256;        // 0..511 chunks of 8 bf16
            const int r   = idx >> 2;              // row 0..127
            const int c8  = idx & 3;               // 8-elem chunk within K panel
            const unsigned la = (unsigned)(size_t)&lA[buf][r * GB_K + c8 * 8];
            const unsigned lb = (unsigned)(size_t)&lB[buf][r * GB_K + c8 * 8];
            async_load_b128(la, (unsigned long long)(size_t)(A + (size_t)(bm + r) * K + k0 + c8 * 8));
            async_load_b128(lb, (unsigned long long)(size_t)(B + (size_t)(bn + r) * K + k0 + c8 * 8));
        }
    };

    stage(0, 0);
    wait_asynccnt0();
    __syncthreads();

    const int nk = K / GB_K;
    for (int kt = 0; kt < nk; ++kt) {
        const int buf = kt & 1;
        if (kt + 1 < nk) {
            stage((kt + 1) * GB_K, buf ^ 1);       // DMA next panel during WMMA
            __builtin_prefetch(B + (size_t)(bn + (tid & 127)) * K + (kt + 2) * GB_K, 0, 1);
        }
#pragma unroll
        for (int mi = 0; mi < 2; ++mi) {
            // A fragment: lane lo = row, chunk0 at k=hi*8, chunk1 at k=16+hi*8.
            const bf16_t* pa = &lA[buf][(wm + mi * 16 + lo) * GB_K + hi * 8];
            bf16x8 a0 = *(const bf16x8*)(pa);
            bf16x8 a1 = *(const bf16x8*)(pa + 16);
            bf16x16 afrag = __builtin_shufflevector(a0, a1,
                0, 1, 2, 3, 4, 5, 6, 7, 8, 9, 10, 11, 12, 13, 14, 15);
#pragma unroll
            for (int ni = 0; ni < 4; ++ni) {
                // B fragment: lane lo = column, chunk0 at k=hi*8, chunk1 at k=16+hi*8.
                const bf16_t* pb = &lB[buf][(wn + ni * 16 + lo) * GB_K + hi * 8];
                bf16x16 bfrag = __builtin_shufflevector(*(const bf16x8*)(pb), *(const bf16x8*)(pb + 16),
                    0, 1, 2, 3, 4, 5, 6, 7, 8, 9, 10, 11, 12, 13, 14, 15);
                acc[mi][ni] = wmma_bf16(afrag, bfrag, acc[mi][ni]);
            }
        }
        wait_asynccnt0();
        __syncthreads();
    }

    // C/D layout: lane 0-15 VGPR v -> M=v; lanes 16-31 -> M=v+8; N=lo.
    if (store_bf16) {
        bf16_t* C = (bf16_t*)Cout;
#pragma unroll
        for (int mi = 0; mi < 2; ++mi)
#pragma unroll
            for (int ni = 0; ni < 4; ++ni)
#pragma unroll
                for (int v = 0; v < 8; ++v)
                    C[(size_t)(bm + wm + mi * 16 + hi * 8 + v) * N + bn + wn + ni * 16 + lo] =
                        (bf16_t)acc[mi][ni][v];
    } else {
        float* C = (float*)Cout;
#pragma unroll
        for (int mi = 0; mi < 2; ++mi)
#pragma unroll
            for (int ni = 0; ni < 4; ++ni)
#pragma unroll
                for (int v = 0; v < 8; ++v)
                    C[(size_t)(bm + wm + mi * 16 + hi * 8 + v) * N + bn + wn + ni * 16 + lo] =
                        acc[mi][ni][v];
    }
}

// ---------------------------------------------------------------------------
// RoPE in-place on bf16 [BSZ][SEQLEN][nheads][64]; optional scale fold
// (Q gets 1/sqrt(HEAD_DIM) folded here so attention loads raw fragments).
// ---------------------------------------------------------------------------
__global__ void rope_bf16_kernel(bf16_t* __restrict__ t, const float* __restrict__ cosv,
                                 const float* __restrict__ sinv, int nheads,
                                 float scale, int total) {
    const int idx = blockIdx.x * blockDim.x + threadIdx.x;
    if (idx >= total) return;
    const int half = HEAD_DIM / 2;
    const int i = idx % half;
    const int h = (idx / half) % nheads;
    const int s = (idx / (half * nheads)) % SEQLEN;
    const int b = idx / (half * nheads * SEQLEN);
    const float c  = cosv[s * half + i];
    const float sn = sinv[s * half + i];
    const size_t base = ((((size_t)b * SEQLEN + s) * nheads) + h) * HEAD_DIM + 2 * i;
    const float xr = (float)t[base], xi = (float)t[base + 1];
    t[base]     = (bf16_t)((xr * c - xi * sn) * scale);
    t[base + 1] = (bf16_t)((xr * sn + xi * c) * scale);
}

// ---------------------------------------------------------------------------
// Flash attention on bf16 Q/K/V. One WG per (64-query tile, head, batch),
// 4 waves each owning 16 rows.  K tile staged by the Tensor Data Mover
// (async fallback), V staged transposed, P reshaped through per-wave LDS.
// ---------------------------------------------------------------------------
__global__ __launch_bounds__(128)
void flash_attn_kernel(const bf16_t* __restrict__ Q, const bf16_t* __restrict__ K,
                       const bf16_t* __restrict__ V, bf16_t* __restrict__ O) {
    __shared__ __align__(32) bf16_t lK[64 * HEAD_DIM];    // [key][d]
    __shared__ __align__(32) bf16_t lVt[HEAD_DIM * 64];   // [d][key]
    __shared__ __align__(32) bf16_t lP[4][16 * 64];       // per-wave P tile

    const int tid  = threadIdx.x;
    const int lane = tid & 31;
    const int wid  = tid >> 5;
    const int hi   = lane >> 4;
    const int lo   = lane & 15;

    const int qb  = blockIdx.x;
    const int h   = blockIdx.y;
    const int b   = blockIdx.z;
    const int kvh = h / N_REP;
    const int qrow0 = qb * 64 + wid * 16;

    // Q tile (16x64) straight into two A fragments (scale pre-folded by RoPE).
    bf16x16 qfrag[2];
    {
        const bf16_t* qp = Q + ((((size_t)b * SEQLEN + qrow0 + lo) * N_HEADS) + h) * HEAD_DIM;
#pragma unroll
        for (int f = 0; f < 2; ++f) {
            bf16x8 a0 = *(const bf16x8*)(qp + f * 32 + hi * 8);
            bf16x8 a1 = *(const bf16x8*)(qp + f * 32 + 16 + hi * 8);
            qfrag[f] = __builtin_shufflevector(a0, a1,
                0, 1, 2, 3, 4, 5, 6, 7, 8, 9, 10, 11, 12, 13, 14, 15);
        }
    }

    floatx8 oacc[4];
#pragma unroll
    for (int i = 0; i < 4; ++i) oacc[i] = (floatx8)0.0f;
    float mrow[8], lrow[8];
#pragma unroll
    for (int v = 0; v < 8; ++v) { mrow[v] = -1e30f; lrow[v] = 0.0f; }

    const int nkb = qb + 1;                    // causal
    for (int jb = 0; jb < nkb; ++jb) {
        const size_t kvbase =
            ((((size_t)b * SEQLEN + jb * 64) * N_KV_HEADS) + kvh) * HEAD_DIM;

        // --- K tile: Tensor Data Mover (wave 0 issues; EXEC-uniform branch) ---
#ifdef HAVE_TDM
        if (wid == 0)
            tdm_load_tile_64x64_bf16((unsigned)(size_t)&lK[0],
                                     (unsigned long long)(size_t)(K + kvbase),
                                     KV_DIM);
#else
#pragma unroll
        for (int it = 0; it < 4; ++it) {
            const int idx = tid + it * 128;    // 0..511 chunks of 8 bf16
            const int r   = idx >> 3;
            const int c8  = idx & 7;
            async_load_b128((unsigned)(size_t)&lK[r * HEAD_DIM + c8 * 8],
                            (unsigned long long)(size_t)(K + kvbase + (size_t)r * KV_DIM + c8 * 8));
        }
#endif
        // --- V tile: transpose-stage [d][key] for contiguous PV B-fragments ---
#pragma unroll
        for (int it = 0; it < 4; ++it) {
            const int idx = tid + it * 128;
            const int r   = idx >> 3;          // key 0..63
            const int c8  = idx & 7;
            bf16x8 vv = *(const bf16x8*)(V + kvbase + (size_t)r * KV_DIM + c8 * 8);
#pragma unroll
            for (int j = 0; j < 8; ++j)
                lVt[(c8 * 8 + j) * 64 + r] = vv[j];
        }
#ifdef HAVE_TDM
        if (wid == 0) __builtin_amdgcn_s_wait_tensorcnt(0);
#else
        wait_asynccnt0();
#endif
        __syncthreads();

        // --- scores = Q * K^T (two 32-wide K steps over head dim) ---
        floatx8 sc[4];
#pragma unroll
        for (int ni = 0; ni < 4; ++ni) sc[ni] = (floatx8)0.0f;
#pragma unroll
        for (int ks = 0; ks < 2; ++ks)
#pragma unroll
            for (int ni = 0; ni < 4; ++ni) {
                const bf16_t* kp = &lK[(ni * 16 + lo) * HEAD_DIM + ks * 32 + hi * 8];
                bf16x16 kf = __builtin_shufflevector(*(const bf16x8*)(kp), *(const bf16x8*)(kp + 16),
                    0, 1, 2, 3, 4, 5, 6, 7, 8, 9, 10, 11, 12, 13, 14, 15);
                sc[ni] = wmma_bf16(qfrag[ks], kf, sc[ni]);
            }

        // --- causal mask + online softmax (16-lane xor reductions stay in the
        //     half-wave that owns rows hi*8+v) ---
        const bool diag = (jb == qb);
        float pv[4][8];
#pragma unroll
        for (int ni = 0; ni < 4; ++ni) {
            const int kcol = jb * 64 + ni * 16 + lo;
#pragma unroll
            for (int v = 0; v < 8; ++v) {
                float s = sc[ni][v];
                if (diag && kcol > (qrow0 + hi * 8 + v)) s = -1e30f;
                pv[ni][v] = s;
            }
        }
#pragma unroll
        for (int v = 0; v < 8; ++v) {
            float m = fmaxf(fmaxf(pv[0][v], pv[1][v]), fmaxf(pv[2][v], pv[3][v]));
#pragma unroll
            for (int off = 1; off < 16; off <<= 1)
                m = fmaxf(m, __shfl_xor(m, off, 32));
            const float mnew  = fmaxf(mrow[v], m);
            const float alpha = __expf(mrow[v] - mnew);
            mrow[v] = mnew;
            float rs = 0.0f;
#pragma unroll
            for (int ni = 0; ni < 4; ++ni) {
                const float p = __expf(pv[ni][v] - mnew);
                pv[ni][v] = p;
                rs += p;
            }
#pragma unroll
            for (int off = 1; off < 16; off <<= 1)
                rs += __shfl_xor(rs, off, 32);
            lrow[v] = lrow[v] * alpha + rs;
#pragma unroll
            for (int ni = 0; ni < 4; ++ni) oacc[ni][v] *= alpha;
        }

        // --- P (C-layout) -> A-layout via per-wave LDS ---
#pragma unroll
        for (int ni = 0; ni < 4; ++ni)
#pragma unroll
            for (int v = 0; v < 8; ++v)
                lP[wid][(hi * 8 + v) * 64 + ni * 16 + lo] = (bf16_t)pv[ni][v];
        asm volatile("s_wait_dscnt 0" ::: "memory");

        // --- O += P * V ---
#pragma unroll
        for (int ks = 0; ks < 2; ++ks) {
            const bf16_t* pp = &lP[wid][lo * 64 + ks * 32 + hi * 8];
            bf16x8 p0 = *(const bf16x8*)(pp);
            bf16x8 p1 = *(const bf16x8*)(pp + 16);
            bf16x16 pfrag = __builtin_shufflevector(p0, p1,
                0, 1, 2, 3, 4, 5, 6, 7, 8, 9, 10, 11, 12, 13, 14, 15);
#pragma unroll
            for (int ni = 0; ni < 4; ++ni) {
                const bf16_t* vp = &lVt[(ni * 16 + lo) * 64 + ks * 32 + hi * 8];
                bf16x16 vf = __builtin_shufflevector(*(const bf16x8*)(vp), *(const bf16x8*)(vp + 16),
                    0, 1, 2, 3, 4, 5, 6, 7, 8, 9, 10, 11, 12, 13, 14, 15);
                oacc[ni] = wmma_bf16(pfrag, vf, oacc[ni]);
            }
        }
        __syncthreads();
    }

    // Normalize, write bf16 attention output [b][s][h][d].
#pragma unroll
    for (int v = 0; v < 8; ++v) {
        const float inv = 1.0f / lrow[v];
        const int qrow = qrow0 + hi * 8 + v;
        bf16_t* op = O + ((((size_t)b * SEQLEN + qrow) * N_HEADS) + h) * HEAD_DIM;
#pragma unroll
        for (int ni = 0; ni < 4; ++ni)
            op[ni * 16 + lo] = (bf16_t)(oacc[ni][v] * inv);
    }
}

// ---------------------------------------------------------------------------
// Launch: bf16 convert -> QKV GEMMs -> RoPE -> flash attention -> out GEMM.
// Inputs: x, freqs_cos, freqs_sin, mask(unused), wq, wk, wv, wo  (all fp32).
// ---------------------------------------------------------------------------
extern "C" void kernel_launch(void* const* d_in, const int* in_sizes, int n_in,
                              void* d_out, int out_size, void* d_ws, size_t ws_size,
                              hipStream_t stream) {
    (void)in_sizes; (void)n_in; (void)out_size; (void)ws_size;
    const float* x  = (const float*)d_in[0];
    const float* fc = (const float*)d_in[1];
    const float* fs = (const float*)d_in[2];
    const float* wq = (const float*)d_in[4];
    const float* wk = (const float*)d_in[5];
    const float* wv = (const float*)d_in[6];
    const float* wo = (const float*)d_in[7];
    float* out = (float*)d_out;

    const int M = BSZ * SEQLEN;                       // 4096 token rows
    bf16_t* p   = (bf16_t*)d_ws;
    bf16_t* Xb  = p; p += (size_t)M * DIM;            // x        (M x 2048)
    bf16_t* Wqb = p; p += (size_t)DIM * DIM;          // wq
    bf16_t* Wkb = p; p += (size_t)KV_DIM * DIM;       // wk
    bf16_t* Wvb = p; p += (size_t)KV_DIM * DIM;       // wv
    bf16_t* Wob = p; p += (size_t)DIM * DIM;          // wo
    bf16_t* Qb  = p; p += (size_t)M * DIM;            // Q        (M x 2048)
    bf16_t* Kb  = p; p += (size_t)M * KV_DIM;         // K        (M x 512)
    bf16_t* Vb  = p; p += (size_t)M * KV_DIM;         // V        (M x 512)
    bf16_t* Ab  = p;                                  // attn out (M x 2048)

    auto cvt = [&](const float* s, bf16_t* d, size_t n) {
        cvt_f32_to_bf16<<<(unsigned)((n / 4 + 255) / 256), 256, 0, stream>>>(s, d, (int)n);
    };
    cvt(x,  Xb,  (size_t)M * DIM);
    cvt(wq, Wqb, (size_t)DIM * DIM);
    cvt(wk, Wkb, (size_t)KV_DIM * DIM);
    cvt(wv, Wvb, (size_t)KV_DIM * DIM);
    cvt(wo, Wob, (size_t)DIM * DIM);

    const dim3 blk(256);
    gemm_bf16<<<dim3(DIM / GB_N, M / GB_M), blk, 0, stream>>>(Xb, Wqb, Qb, M, DIM, DIM, 1);
    gemm_bf16<<<dim3(KV_DIM / GB_N, M / GB_M), blk, 0, stream>>>(Xb, Wkb, Kb, M, KV_DIM, DIM, 1);
    gemm_bf16<<<dim3(KV_DIM / GB_N, M / GB_M), blk, 0, stream>>>(Xb, Wvb, Vb, M, KV_DIM, DIM, 1);

    const int nq = M * N_HEADS * (HEAD_DIM / 2);
    rope_bf16_kernel<<<(nq + 255) / 256, 256, 0, stream>>>(Qb, fc, fs, N_HEADS, 0.125f, nq);
    const int nk = M * N_KV_HEADS * (HEAD_DIM / 2);
    rope_bf16_kernel<<<(nk + 255) / 256, 256, 0, stream>>>(Kb, fc, fs, N_KV_HEADS, 1.0f, nk);

    flash_attn_kernel<<<dim3(SEQLEN / 64, N_HEADS, BSZ), dim3(128), 0, stream>>>(Qb, Kb, Vb, Ab);

    gemm_bf16<<<dim3(DIM / GB_N, M / GB_M), blk, 0, stream>>>(Ab, Wob, out, M, DIM, DIM, 0);
}
